// CGCNNLayer_84335977824815
// MI455X (gfx1250) — matrix operand order repaired and checked by reading, not compiled, not run on hardware
//
#include <hip/hip_runtime.h>
#include <hip/hip_bf16.h>

typedef __attribute__((ext_vector_type(16))) __bf16 v16bf;
typedef __attribute__((ext_vector_type(8)))  float  v8f;

#define HID 128
#define K3 384              // 3 * HID
#define TILE_E 64
#define LDS_STRIDE 392      // K3 + 8 pad (halfwords) -> rotates LDS banks per row

__device__ __forceinline__ unsigned short f2bf(float f) {
  unsigned u = __float_as_uint(f);
  u += 0x7FFFu + ((u >> 16) & 1u);   // round-to-nearest-even
  return (unsigned short)(u >> 16);
}

union V16U { uint4 q[2]; v16bf v; unsigned short s[16]; };

__global__ void cgcnn_zero_out(float* __restrict__ o, long long n) {
  long long i = (long long)blockIdx.x * blockDim.x + threadIdx.x;
  if (i < n) o[i] = 0.0f;
}

// Convert W_weight[384x128], W_gate[384x128] (f32, K-major) into bf16
// transposed Bt[256][384]: row n<128 -> W_weight column n; n>=128 -> W_gate.
__global__ void cgcnn_cvt_weights(const float* __restrict__ Ww,
                                  const float* __restrict__ Wg,
                                  unsigned short* __restrict__ Bt) {
  int tid = blockIdx.x * blockDim.x + threadIdx.x;
  if (tid >= 256 * K3) return;
  int n = tid / K3;
  int k = tid - n * K3;
  const float* W = (n < HID) ? Ww : Wg;
  int nn = n & (HID - 1);
  Bt[tid] = f2bf(W[k * HID + nn]);
}

template <bool FAST>
__global__ __launch_bounds__(256, 1)
void cgcnn_edge_kernel(const float* __restrict__ h,
                       const long long* __restrict__ eidx,
                       const float* __restrict__ eattr,
                       const float* __restrict__ Ww,
                       const float* __restrict__ bw,
                       const float* __restrict__ Wg,
                       const float* __restrict__ bg,
                       const unsigned short* __restrict__ Bt,
                       float* __restrict__ out,
                       int E) {
  __shared__ unsigned short Alds[TILE_E * LDS_STRIDE];
  __shared__ int rowS[TILE_E];
  __shared__ int colS[TILE_E];

  const int tid = threadIdx.x;
  const long long e0 = (long long)blockIdx.x * TILE_E;

  // cache edge endpoint indices for this tile
  if (tid < TILE_E) {
    long long e = e0 + tid;
    long long r = 0, c = 0;
    if (e < (long long)E) { r = eidx[e]; c = eidx[(long long)E + e]; }
    rowS[tid] = (int)r;
    colS[tid] = (int)c;
  }
  __syncthreads();

  // ---- gather phase: build A tile [64 edges x 384] in bf16 ----
  // 64 rows * 96 float4-chunks; 24 iterations per thread, uniform trip count.
  for (int cidx = tid; cidx < TILE_E * 96; cidx += 256) {
    int row = cidx / 96;
    int q   = cidx - row * 96;
    int seg = q >> 5;        // 0: h[row], 1: h[col], 2: edge_attr
    int qi  = q & 31;        // float4 index within 128-float segment
    long long e = e0 + row;
    float4 v = make_float4(0.f, 0.f, 0.f, 0.f);
    if (e < (long long)E) {
      const float* src;
      if (seg == 0)      src = h + (long long)rowS[row] * HID;
      else if (seg == 1) src = h + (long long)colS[row] * HID;
      else               src = eattr + e * HID;
      v = *(const float4*)(src + qi * 4);
    }
    unsigned lo = (unsigned)f2bf(v.x) | ((unsigned)f2bf(v.y) << 16);
    unsigned hi = (unsigned)f2bf(v.z) | ((unsigned)f2bf(v.w) << 16);
    int off = seg * HID + qi * 4;
    *(uint2*)&Alds[row * LDS_STRIDE + off] = make_uint2(lo, hi);
  }
  __syncthreads();

  // ---- compute phase: each wave owns a 16-column strip of both outputs ----
  const int lane  = tid & 31;
  const int wave  = tid >> 5;          // 0..7 -> column tile
  const int nl    = lane & 15;
  const int khalf = (lane >> 4) & 1;   // lane half selects K sub-pattern
  const int ncol  = wave * 16 + nl;

  const float bwv = bw[ncol];
  const float bgv = bg[ncol];
  v8f accM[4], accG[4];
#pragma unroll
  for (int s = 0; s < 4; ++s)
#pragma unroll
    for (int r = 0; r < 8; ++r) { accM[s][r] = bwv; accG[s][r] = bgv; }

  const unsigned short* BtM = Bt + (size_t)ncol * K3;
  const unsigned short* BtG = Bt + (size_t)(HID + ncol) * K3;

  for (int kc = 0; kc < K3; kc += 32) {
    const int k0a = kc + khalf * 16;   // A: contiguous K per lane-half
    const int k0b = kc + khalf * 16;   // B: contiguous K per lane-half

    V16U bM, bG;
    if (FAST) {
      bM.q[0] = *(const uint4*)(BtM + k0b);
      bM.q[1] = *(const uint4*)(BtM + k0b + 8);
      bG.q[0] = *(const uint4*)(BtG + k0b);
      bG.q[1] = *(const uint4*)(BtG + k0b + 8);
    } else {
#pragma unroll
      for (int i = 0; i < 16; ++i) {
        int k = k0b + i;
        bM.s[i] = f2bf(Ww[k * HID + ncol]);
        bG.s[i] = f2bf(Wg[k * HID + ncol]);
      }
    }

#pragma unroll
    for (int s = 0; s < 4; ++s) {
      const int arow = s * 16 + nl;
      V16U a;
      a.q[0] = *(const uint4*)&Alds[arow * LDS_STRIDE + k0a];
      a.q[1] = *(const uint4*)&Alds[arow * LDS_STRIDE + k0a + 8];
      accM[s] = __builtin_amdgcn_wmma_f32_16x16x32_bf16(
          false, a.v, false, bM.v, (short)0, accM[s], false, false);
      accG[s] = __builtin_amdgcn_wmma_f32_16x16x32_bf16(
          false, a.v, false, bG.v, (short)0, accG[s], false, false);
    }
  }

  // ---- epilogue: relu(m) * sigmoid(g), scatter-add to source node rows ----
#pragma unroll
  for (int s = 0; s < 4; ++s) {
#pragma unroll
    for (int r = 0; r < 8; ++r) {
      const int el = s * 16 + khalf * 8 + r;   // edge index within tile
      const long long e = e0 + el;
      if (e < (long long)E) {
        const float m = accM[s][r];
        const float g = accG[s][r];
        const float val = fmaxf(m, 0.0f) * (1.0f / (1.0f + __expf(-g)));
        atomicAdd(out + (long long)rowS[el] * HID + ncol, val);
      }
    }
  }
}

extern "C" void kernel_launch(void* const* d_in, const int* in_sizes, int n_in,
                              void* d_out, int out_size, void* d_ws, size_t ws_size,
                              hipStream_t stream) {
  const float*     h     = (const float*)d_in[0];
  const long long* eidx  = (const long long*)d_in[1];   // int64 per reference
  const float*     eattr = (const float*)d_in[2];
  const float*     Ww    = (const float*)d_in[3];
  const float*     bw    = (const float*)d_in[4];
  const float*     Wg    = (const float*)d_in[5];
  const float*     bg    = (const float*)d_in[6];
  float*           out   = (float*)d_out;

  const int E = in_sizes[1] / 2;                 // edge_indices is [2, E]

  // zero the output (harness poisons d_out); graph-capture-safe kernel memset
  {
    long long n = (long long)out_size;
    int blocks = (int)((n + 255) / 256);
    cgcnn_zero_out<<<blocks, 256, 0, stream>>>(out, n);
  }

  const size_t btBytes = (size_t)256 * K3 * sizeof(unsigned short);
  const bool fast = (ws_size >= btBytes);
  unsigned short* Bt = (unsigned short*)d_ws;

  if (fast) {
    int total = 256 * K3;
    cgcnn_cvt_weights<<<(total + 255) / 256, 256, 0, stream>>>(Ww, Wg, Bt);
  }

  const int nTiles = (E + TILE_E - 1) / TILE_E;  // 12500 for E=800000
  if (fast) {
    cgcnn_edge_kernel<true><<<nTiles, 256, 0, stream>>>(
        h, eidx, eattr, Ww, bw, Wg, bg, Bt, out, E);
  } else {
    cgcnn_edge_kernel<false><<<nTiles, 256, 0, stream>>>(
        h, eidx, eattr, Ww, bw, Wg, bg, Bt, out, E);
  }
}
